// ExpireSpanTransformerXL_44452911514182
// MI455X (gfx1250) — matrix operand order repaired and checked by reading, not compiled, not run on hardware
//
#include <hip/hip_runtime.h>
#include <math.h>

// ---- model constants (from reference) ----
#define VOCABN 32000
#define DIMN   1024
#define DEPTHN 4
#define SEQN   512
#define HEADSN 16
#define MAXMEMN 2048
#define MTN    (MAXMEMN + SEQN)   // 2560
#define RAMPF  128.0f
#define COEFF  1e-6f
#define BBN    4
#define DHN    64                 // DIM/HEADS

typedef __bf16 v16bf __attribute__((ext_vector_type(16)));
typedef float  v8f   __attribute__((ext_vector_type(8)));
typedef int    v4i   __attribute__((ext_vector_type(4)));

union Frag {
    v16bf v;
    uint4 u[2];
};

// ---- CDNA5 async global->LDS path (guarded; sync fallback keeps build safe) ----
#if defined(__AMDGCN__) && __has_builtin(__builtin_amdgcn_global_load_async_to_lds_b128)
#define HAVE_ASYNC 1
#else
#define HAVE_ASYNC 0
#endif

__device__ __forceinline__ void async_copy16(const __bf16* g, __bf16* lds) {
#if HAVE_ASYNC
    __builtin_amdgcn_global_load_async_to_lds_b128((v4i*)g, (v4i*)lds, 0, 0);
#else
    *(uint4*)lds = *(const uint4*)g;
#endif
}

__device__ __forceinline__ void wait_async_le3() {
#if HAVE_ASYNC
#if __has_builtin(__builtin_amdgcn_s_wait_asynccnt)
    __builtin_amdgcn_s_wait_asynccnt(3);
#else
    asm volatile("s_wait_asynccnt 0x3" ::: "memory");
#endif
#endif
}
__device__ __forceinline__ void wait_async_le0() {
#if HAVE_ASYNC
#if __has_builtin(__builtin_amdgcn_s_wait_asynccnt)
    __builtin_amdgcn_s_wait_asynccnt(0);
#else
    asm volatile("s_wait_asynccnt 0x0" ::: "memory");
#endif
#endif
}

// ---------------- embedding gather (f32) ----------------
__global__ void embed_kernel(const int* __restrict__ tokens,
                             const float* __restrict__ emb,
                             float* __restrict__ x) {
    size_t idx = (size_t)blockIdx.x * 256 + threadIdx.x;
    if (idx >= (size_t)BBN * SEQN * DIMN) return;
    size_t row = idx / DIMN; int c = (int)(idx % DIMN);
    int tok = tokens[row];
    x[idx] = emb[(size_t)tok * DIMN + c];
}

// ---------------- plain f32 -> bf16 convert ----------------
__global__ void cvt_kernel(const float* __restrict__ in, __bf16* __restrict__ out, size_t n) {
    size_t idx = (size_t)blockIdx.x * 256 + threadIdx.x;
    if (idx < n) out[idx] = (__bf16)in[idx];
}

// ---------------- weight transpose+convert: W(KxN f32) -> Wt(NxK bf16) ----------------
__global__ __launch_bounds__(256)
void wt_kernel(const float* __restrict__ W, __bf16* __restrict__ Wt, int K, int N) {
    __shared__ float t[32][33];
    int kb = blockIdx.y * 32, nb = blockIdx.x * 32;
    int tx = threadIdx.x & 31, ty = threadIdx.x >> 5;   // 32 x 8
    for (int i = ty; i < 32; i += 8)
        t[i][tx] = W[(size_t)(kb + i) * N + nb + tx];
    __syncthreads();
    for (int i = ty; i < 32; i += 8)
        Wt[(size_t)(nb + i) * K + kb + tx] = (__bf16)t[tx][i];
}

// ---------------- layer norm, bf16 output ----------------
__global__ void ln_kernel(const float* __restrict__ x,
                          const float* __restrict__ g,
                          const float* __restrict__ b,
                          __bf16* __restrict__ y) {
    __shared__ float red[256];
    int row = blockIdx.x, tid = threadIdx.x;
    const float* xr = x + (size_t)row * DIMN;
    float s = 0.f;
    for (int i = tid; i < DIMN; i += 256) s += xr[i];
    red[tid] = s; __syncthreads();
    for (int o = 128; o > 0; o >>= 1) { if (tid < o) red[tid] += red[tid + o]; __syncthreads(); }
    float mu = red[0] * (1.0f / DIMN);
    __syncthreads();
    float v = 0.f;
    for (int i = tid; i < DIMN; i += 256) { float d = xr[i] - mu; v += d * d; }
    red[tid] = v; __syncthreads();
    for (int o = 128; o > 0; o >>= 1) { if (tid < o) red[tid] += red[tid + o]; __syncthreads(); }
    float inv = rsqrtf(red[0] * (1.0f / DIMN) + 1e-5f);
    __bf16* yr = y + (size_t)row * DIMN;
    for (int i = tid; i < DIMN; i += 256) yr[i] = (__bf16)((xr[i] - mu) * inv * g[i] + b[i]);
}

// ---------------- expire spans: sigmoid(mem . W + b) * M ----------------
__global__ void exps_kernel(const float* __restrict__ mem,
                            const float* __restrict__ ew,
                            const float* __restrict__ ebp, int d,
                            float* __restrict__ exps) {
    __shared__ float red[256];
    int row = blockIdx.x, tid = threadIdx.x;
    const float* mr = mem + (size_t)row * DIMN;
    float s = 0.f;
    for (int i = tid; i < DIMN; i += 256) s += mr[i] * ew[i];
    red[tid] = s; __syncthreads();
    for (int o = 128; o > 0; o >>= 1) { if (tid < o) red[tid] += red[tid + o]; __syncthreads(); }
    if (tid == 0) {
        float z = red[0] + ebp[d];
        exps[row] = (1.0f / (1.0f + expf(-z))) * (float)MAXMEMN;
    }
}

// ---------------- emask + aux (single block, deterministic) ----------------
__global__ void emask_aux_kernel(const float* __restrict__ exps,
                                 const int* __restrict__ times_d,
                                 float* __restrict__ emask,
                                 float* __restrict__ aux) {
    __shared__ float red[256];
    int tid = threadIdx.x;
    float s = 0.f;
    for (int i = tid; i < BBN * MAXMEMN; i += 256) s += exps[i];
    for (int i = tid; i < BBN * MTN; i += 256) {
        int b = i / MTN, j = i % MTN;
        float em = 1.0f;
        if (j < MAXMEMN) {
            float r = exps[b * MAXMEMN + j] - (float)times_d[j];
            em = fminf(fmaxf(r * (1.0f / RAMPF) + 1.0f, 0.0f), 1.0f);
        }
        emask[i] = em;
    }
    red[tid] = s; __syncthreads();
    for (int o = 128; o > 0; o >>= 1) { if (tid < o) red[tid] += red[tid + o]; __syncthreads(); }
    if (tid == 0) aux[0] += red[0] * (1.0f / (float)SEQN) * COEFF;
}

__global__ void zero1_kernel(float* p) { if (threadIdx.x == 0) p[0] = 0.0f; }

// ---------------- memory shift: concat(mem, hidden)[:, -M:] ----------------
__global__ void memup_kernel(const float* __restrict__ mem,
                             const float* __restrict__ hidden,
                             float* __restrict__ omem) {
    size_t idx = (size_t)blockIdx.x * 256 + threadIdx.x;
    if (idx >= (size_t)BBN * MAXMEMN * DIMN) return;
    size_t b = idx / ((size_t)MAXMEMN * DIMN);
    size_t rem = idx % ((size_t)MAXMEMN * DIMN);
    int j = (int)(rem / DIMN), c = (int)(rem % DIMN);
    float v;
    if (j < MAXMEMN - SEQN) v = mem[((size_t)b * MAXMEMN + j + SEQN) * DIMN + c];
    else                    v = hidden[((size_t)b * SEQN + (j - (MAXMEMN - SEQN))) * DIMN + c];
    omem[idx] = v;
}

__global__ void timesup_kernel(const int* __restrict__ t, float* __restrict__ ot) {
    int j = blockIdx.x * 256 + threadIdx.x;
    if (j >= MAXMEMN) return;
    ot[j] = (j < MAXMEMN - SEQN) ? (float)(t[j + SEQN] + SEQN) : (float)(MAXMEMN - 1 - j);
}

// ---------------- ctx = concat(mem (f32), xn (bf16)) -> bf16 ----------------
__global__ void ctx_kernel(const float* __restrict__ mem,
                           const __bf16* __restrict__ xn,
                           __bf16* __restrict__ ctx) {
    size_t idx = (size_t)blockIdx.x * 256 + threadIdx.x;
    if (idx >= (size_t)BBN * MTN * DIMN) return;
    size_t b = idx / ((size_t)MTN * DIMN);
    size_t rem = idx % ((size_t)MTN * DIMN);
    int r = (int)(rem / DIMN), c = (int)(rem % DIMN);
    ctx[idx] = (r < MAXMEMN) ? (__bf16)mem[((size_t)b * MAXMEMN + r) * DIMN + c]
                             : xn[((size_t)b * SEQN + (r - MAXMEMN)) * DIMN + c];
}

// ---------------- V transpose per (b,h): kv_bf [j][2*DIM] -> vT [b*H+h][dh][j] -------
__global__ __launch_bounds__(256)
void vtr_kernel(const __bf16* __restrict__ kvb, __bf16* __restrict__ vT) {
    __shared__ __bf16 t[32][DHN + 2];
    int jb = blockIdx.x * 32;
    int bh = blockIdx.y;
    int b = bh >> 4, h = bh & 15;
    int tx = threadIdx.x & 63, ty = threadIdx.x >> 6;    // 64 x 4
    for (int i = ty; i < 32; i += 4)
        t[i][tx] = kvb[((size_t)b * MTN + jb + i) * (2 * DIMN) + DIMN + h * DHN + tx];
    __syncthreads();
    int tx2 = threadIdx.x & 31, ty2 = threadIdx.x >> 5;  // 32 x 8
    for (int i = ty2; i < DHN; i += 8)
        vT[((size_t)bh * DHN + i) * MTN + jb + tx2] = t[tx2][i];
}

// ---------------- bf16 WMMA GEMM: C = A(MxK) * Bt(NxK)^T, 128x64 block tile -------
// FLAGS: bit0 = bias, bit1 = gelu, bit2 = residual, bit3 = bf16 output
template <int FLAGS>
__global__ __launch_bounds__(256)
void gemm_kernel(const __bf16* __restrict__ A,    // (Mr x K) row-major bf16
                 const __bf16* __restrict__ Bt,   // (N  x K) row-major bf16
                 const float* __restrict__ bias,
                 const float* __restrict__ Res,
                 float* __restrict__ Cf, __bf16* __restrict__ Ch,
                 int Mr, int N, int K) {
    __shared__ __align__(16) __bf16 Atile[2][128 * 32];
    __shared__ __align__(16) __bf16 Btile[2][64 * 32];
    int tid = threadIdx.x, lane = tid & 31, wid = tid >> 5;
    int half = lane >> 4, r = lane & 15;
    int wm = wid & 3, wn = wid >> 2;                 // 4x2 waves, each 32x32 output
    int rowblock = blockIdx.y * 128;
    int colblock = blockIdx.x * 64;

    v8f acc[2][2];
    v8f vz = {0, 0, 0, 0, 0, 0, 0, 0};
    acc[0][0] = vz; acc[0][1] = vz; acc[1][0] = vz; acc[1][1] = vz;

    // staging: per wave, 2 b128 for A (16 rows) + 1 b128 for B (8 rows)
    int srow = lane >> 2;
    int scol = (lane & 3) * 8;                       // bf16 element offset (16B chunk)
    auto stage = [&](int buf, int kt) {
        int ar = wid * 16 + srow;
        async_copy16(A + (size_t)(rowblock + ar) * K + kt + scol,
                     &Atile[buf][ar * 32 + scol]);
        async_copy16(A + (size_t)(rowblock + ar + 8) * K + kt + scol,
                     &Atile[buf][(ar + 8) * 32 + scol]);
        int br = wid * 8 + srow;
        async_copy16(Bt + (size_t)(colblock + br) * K + kt + scol,
                     &Btile[buf][br * 32 + scol]);
    };
    auto compute = [&](int buf) {
        Frag a[2], b[2];
        #pragma unroll
        for (int i = 0; i < 2; ++i) {
            const __bf16* ar = &Atile[buf][(wm * 32 + i * 16 + r) * 32];
            a[i].u[0] = *(const uint4*)(ar + half * 16);
            a[i].u[1] = *(const uint4*)(ar + half * 16 + 8);
            const __bf16* br = &Btile[buf][(wn * 32 + i * 16 + r) * 32 + half * 16];
            b[i].u[0] = ((const uint4*)br)[0];
            b[i].u[1] = ((const uint4*)br)[1];
        }
        #pragma unroll
        for (int i = 0; i < 2; ++i)
            #pragma unroll
            for (int j = 0; j < 2; ++j)
                acc[i][j] = __builtin_amdgcn_wmma_f32_16x16x32_bf16(
                    false, a[i].v, false, b[j].v, (short)0, acc[i][j], false, false);
    };

    stage(0, 0);
    int buf = 0;
    for (int kt = 0; kt < K; kt += 32) {
        int nxt = kt + 32;
        if (nxt < K) { stage(buf ^ 1, nxt); wait_async_le3(); }
        else         { wait_async_le0(); }
        __syncthreads();
        compute(buf);
        __syncthreads();
        buf ^= 1;
    }

    #pragma unroll
    for (int i = 0; i < 2; ++i) {
        #pragma unroll
        for (int j = 0; j < 2; ++j) {
            int gcol = colblock + wn * 32 + j * 16 + r;
            float bv = (FLAGS & 1) ? bias[gcol] : 0.0f;
            #pragma unroll
            for (int rr = 0; rr < 8; ++rr) {
                int grow = rowblock + wm * 32 + i * 16 + rr + 8 * half;
                float v = acc[i][j][rr] + bv;
                if (FLAGS & 2) v = 0.5f * v * (1.0f + erff(v * 0.70710678118f));
                if (FLAGS & 4) v += Res[(size_t)grow * N + gcol];
                if (FLAGS & 8) Ch[(size_t)grow * N + gcol] = (__bf16)v;
                else           Cf[(size_t)grow * N + gcol] = v;
            }
        }
    }
}

// ---------------- flash attention, 1 wave / 16-query tile / (b,h) ----------------
__global__ __launch_bounds__(32)
void attn_kernel(const __bf16* __restrict__ qb,     // (B, SEQ, DIM) bf16
                 const __bf16* __restrict__ kvb,    // (B, MT, 2*DIM) bf16
                 const __bf16* __restrict__ vT,     // (B*H, DH, MT) bf16
                 const float* __restrict__ emask,   // (B, MT)
                 __bf16* __restrict__ outb) {       // (B, SEQ, DIM) bf16
    __shared__ __align__(16) __bf16 Plds[16 * 32];
    int lane = threadIdx.x & 31;
    int half = lane >> 4, r = lane & 15;
    int qbase = blockIdx.x * 16;
    int h = blockIdx.y, b = blockIdx.z;

    Frag qa[2];
    const __bf16* qrow = qb + ((size_t)(b * SEQN + qbase + r)) * DIMN + h * DHN;
    #pragma unroll
    for (int c = 0; c < 2; ++c) {
        qa[c].u[0] = *(const uint4*)(qrow + c * 32 + half * 16);
        qa[c].u[1] = *(const uint4*)(qrow + c * 32 + half * 16 + 8);
    }

    float mrow[8], lrow[8];
    v8f acc[4];
    v8f vz = {0, 0, 0, 0, 0, 0, 0, 0};
    #pragma unroll
    for (int i = 0; i < 8; ++i) { mrow[i] = -1e30f; lrow[i] = 0.0f; }
    #pragma unroll
    for (int c = 0; c < 4; ++c) acc[c] = vz;

    const size_t kvbase = (size_t)b * MTN * (2 * DIMN);
    const size_t vbase = ((size_t)(b * HEADSN + h)) * DHN * MTN;
    int jend = qbase + 16 + MAXMEMN;

    for (int jt = 0; jt < jend; jt += 32) {
        v8f s[2];
        #pragma unroll
        for (int t = 0; t < 2; ++t) {
            int jb = jt + t * 16;
            v8f sc = vz;
            #pragma unroll
            for (int c = 0; c < 2; ++c) {
                Frag kb;
                const __bf16* kp = kvb + kvbase + (size_t)(jb + r) * (2 * DIMN)
                                   + h * DHN + c * 32 + half * 16;
                kb.u[0] = ((const uint4*)kp)[0];
                kb.u[1] = ((const uint4*)kp)[1];
                sc = __builtin_amdgcn_wmma_f32_16x16x32_bf16(
                    false, qa[c].v, false, kb.v, (short)0, sc, false, false);
            }
            s[t] = sc;
        }
        float em0 = emask[(size_t)b * MTN + jt + r];
        float em1 = emask[(size_t)b * MTN + jt + 16 + r];
        #pragma unroll
        for (int rr = 0; rr < 8; ++rr) {
            int qi = qbase + rr + 8 * half;
            float a0 = s[0][rr] * 0.125f;              // DH^-0.5
            float a1 = s[1][rr] * 0.125f;
            if (jt + r      > qi + MAXMEMN) a0 = -1e30f;
            if (jt + 16 + r > qi + MAXMEMN) a1 = -1e30f;
            float mv = fmaxf(a0, a1);
            #pragma unroll
            for (int msk = 1; msk < 16; msk <<= 1) mv = fmaxf(mv, __shfl_xor(mv, msk, 32));
            float mnew = fmaxf(mrow[rr], mv);
            float alpha = expf(mrow[rr] - mnew);
            float p0 = expf(a0 - mnew);
            float p1 = expf(a1 - mnew);
            float rs = p0 + p1;
            #pragma unroll
            for (int msk = 1; msk < 16; msk <<= 1) rs += __shfl_xor(rs, msk, 32);
            lrow[rr] = lrow[rr] * alpha + rs;          // denominator: raw softmax
            mrow[rr] = mnew;
            #pragma unroll
            for (int c = 0; c < 4; ++c) acc[c][rr] *= alpha;
            int row = rr + 8 * half;
            Plds[row * 32 + r]      = (__bf16)(p0 * em0);   // numerator gets emask
            Plds[row * 32 + 16 + r] = (__bf16)(p1 * em1);
        }
        __syncthreads();
        Frag pa;
        {
            const __bf16* prow = &Plds[r * 32];
            pa.u[0] = *(const uint4*)(prow + half * 16);
            pa.u[1] = *(const uint4*)(prow + half * 16 + 8);
        }
        #pragma unroll
        for (int c = 0; c < 4; ++c) {
            Frag vb;
            const __bf16* vp = vT + vbase + (size_t)(c * 16 + r) * MTN + jt + half * 16;
            vb.u[0] = ((const uint4*)vp)[0];
            vb.u[1] = ((const uint4*)vp)[1];
            acc[c] = __builtin_amdgcn_wmma_f32_16x16x32_bf16(
                false, pa.v, false, vb.v, (short)0, acc[c], false, false);
        }
        __syncthreads();
    }
    #pragma unroll
    for (int rr = 0; rr < 8; ++rr) {
        float inv = 1.0f / lrow[rr];
        int row = qbase + rr + 8 * half;
        #pragma unroll
        for (int c = 0; c < 4; ++c)
            outb[((size_t)(b * SEQN + row)) * DIMN + h * DHN + c * 16 + r] =
                (__bf16)(acc[c][rr] * inv);
    }
}

// =======================================================================
extern "C" void kernel_launch(void* const* d_in, const int* in_sizes, int n_in,
                              void* d_out, int out_size, void* d_ws, size_t ws_size,
                              hipStream_t stream) {
    const int*   tokens   = (const int*)  d_in[0];
    const float* mems     = (const float*)d_in[1];
    const int*   times    = (const int*)  d_in[2];
    const float* tok_emb  = (const float*)d_in[3];
    const float* expire_W = (const float*)d_in[4];
    const float* expire_b = (const float*)d_in[5];
    const float* ln1_g    = (const float*)d_in[6];
    const float* ln1_b    = (const float*)d_in[7];
    const float* Wq       = (const float*)d_in[8];
    const float* Wkv      = (const float*)d_in[9];
    const float* Wo       = (const float*)d_in[10];
    const float* bo       = (const float*)d_in[11];
    const float* ln2_g    = (const float*)d_in[12];
    const float* ln2_b    = (const float*)d_in[13];
    const float* W1       = (const float*)d_in[14];
    const float* b1       = (const float*)d_in[15];
    const float* W2       = (const float*)d_in[16];
    const float* b2       = (const float*)d_in[17];
    const float* logits_W = (const float*)d_in[18];
    const float* logits_b = (const float*)d_in[19];

    float* out_logits = (float*)d_out;
    float* out_mems   = out_logits + (size_t)BBN * SEQN * VOCABN;
    float* out_times  = out_mems + (size_t)DEPTHN * BBN * MAXMEMN * DIMN;
    float* out_aux    = out_times + (size_t)DEPTHN * MAXMEMN;

    // ---- workspace carve-up (256B aligned) ----
    char* wsb = (char*)d_ws;
    size_t off = 0;
    auto alloc = [&](size_t bytes) {
        void* p = wsb + off;
        off = (off + bytes + 255) & ~(size_t)255;
        return p;
    };
    const size_t XSZ = (size_t)BBN * SEQN * DIMN;           // 2,097,152
    float*  xA    = (float*)alloc(XSZ * 4);
    float*  xB    = (float*)alloc(XSZ * 4);
    float*  exps  = (float*)alloc((size_t)BBN * MAXMEMN * 4);
    float*  emsk  = (float*)alloc((size_t)BBN * MTN * 4);
    __bf16* xn_bf = (__bf16*)alloc(XSZ * 2);
    __bf16* q_bf  = (__bf16*)alloc(XSZ * 2);
    __bf16* atn_bf= (__bf16*)alloc(XSZ * 2);
    __bf16* xA_bf = (__bf16*)alloc(XSZ * 2);
    __bf16* ctx_bf= (__bf16*)alloc((size_t)BBN * MTN * DIMN * 2);
    __bf16* kv_bf = (__bf16*)alloc((size_t)BBN * MTN * 2 * DIMN * 2);
    __bf16* vT_bf = (__bf16*)alloc((size_t)BBN * HEADSN * DHN * MTN * 2);
    __bf16* ffh_bf= (__bf16*)alloc((size_t)BBN * SEQN * 4 * DIMN * 2);
    __bf16* wT    = (__bf16*)alloc((size_t)4 * DIMN * DIMN * 2);       // max W1/W2
    __bf16* lWt   = (__bf16*)alloc((size_t)DIMN * VOCABN * 2);

    const int ROWS = BBN * SEQN;                            // 2048

    zero1_kernel<<<1, 32, 0, stream>>>(out_aux);
    embed_kernel<<<(unsigned)((XSZ + 255) / 256), 256, 0, stream>>>(tokens, tok_emb, xA);

    for (int d = 0; d < DEPTHN; ++d) {
        const float* mem_d = mems + (size_t)d * BBN * MAXMEMN * DIMN;
        const int*   tim_d = times + (size_t)d * MAXMEMN;

        exps_kernel<<<BBN * MAXMEMN, 256, 0, stream>>>(mem_d, expire_W + (size_t)d * DIMN,
                                                       expire_b, d, exps);
        emask_aux_kernel<<<1, 256, 0, stream>>>(exps, tim_d, emsk, out_aux);

        memup_kernel<<<(unsigned)(((size_t)BBN * MAXMEMN * DIMN + 255) / 256), 256, 0, stream>>>(
            mem_d, xA, out_mems + (size_t)d * BBN * MAXMEMN * DIMN);
        timesup_kernel<<<(MAXMEMN + 255) / 256, 256, 0, stream>>>(
            tim_d, out_times + (size_t)d * MAXMEMN);

        // --- attention ---
        ln_kernel<<<ROWS, 256, 0, stream>>>(xA, ln1_g + (size_t)d * DIMN,
                                            ln1_b + (size_t)d * DIMN, xn_bf);
        wt_kernel<<<dim3(DIMN / 32, DIMN / 32), 256, 0, stream>>>(
            Wq + (size_t)d * DIMN * DIMN, wT, DIMN, DIMN);
        gemm_kernel<8><<<dim3(DIMN / 64, ROWS / 128), 256, 0, stream>>>(
            xn_bf, wT, nullptr, nullptr, nullptr, q_bf, ROWS, DIMN, DIMN);
        ctx_kernel<<<(unsigned)(((size_t)BBN * MTN * DIMN + 255) / 256), 256, 0, stream>>>(
            mem_d, xn_bf, ctx_bf);
        wt_kernel<<<dim3(2 * DIMN / 32, DIMN / 32), 256, 0, stream>>>(
            Wkv + (size_t)d * DIMN * 2 * DIMN, wT, DIMN, 2 * DIMN);
        gemm_kernel<8><<<dim3(2 * DIMN / 64, BBN * MTN / 128), 256, 0, stream>>>(
            ctx_bf, wT, nullptr, nullptr, nullptr, kv_bf, BBN * MTN, 2 * DIMN, DIMN);
        vtr_kernel<<<dim3(MTN / 32, BBN * HEADSN), 256, 0, stream>>>(kv_bf, vT_bf);
        attn_kernel<<<dim3(SEQN / 16, HEADSN, BBN), 32, 0, stream>>>(
            q_bf, kv_bf, vT_bf, emsk, atn_bf);
        wt_kernel<<<dim3(DIMN / 32, DIMN / 32), 256, 0, stream>>>(
            Wo + (size_t)d * DIMN * DIMN, wT, DIMN, DIMN);
        gemm_kernel<5><<<dim3(DIMN / 64, ROWS / 128), 256, 0, stream>>>(
            atn_bf, wT, bo + (size_t)d * DIMN, xA, xB, nullptr, ROWS, DIMN, DIMN);

        // --- FFN ---
        ln_kernel<<<ROWS, 256, 0, stream>>>(xB, ln2_g + (size_t)d * DIMN,
                                            ln2_b + (size_t)d * DIMN, xn_bf);
        wt_kernel<<<dim3(4 * DIMN / 32, DIMN / 32), 256, 0, stream>>>(
            W1 + (size_t)d * DIMN * 4 * DIMN, wT, DIMN, 4 * DIMN);
        gemm_kernel<11><<<dim3(4 * DIMN / 64, ROWS / 128), 256, 0, stream>>>(
            xn_bf, wT, b1 + (size_t)d * 4 * DIMN, nullptr, nullptr, ffh_bf,
            ROWS, 4 * DIMN, DIMN);
        wt_kernel<<<dim3(DIMN / 32, 4 * DIMN / 32), 256, 0, stream>>>(
            W2 + (size_t)d * 4 * DIMN * DIMN, wT, 4 * DIMN, DIMN);
        gemm_kernel<5><<<dim3(DIMN / 64, ROWS / 128), 256, 0, stream>>>(
            ffh_bf, wT, b2 + (size_t)d * DIMN, xB, xA, nullptr, ROWS, DIMN, 4 * DIMN);
    }

    // --- logits ---
    cvt_kernel<<<(unsigned)((XSZ + 255) / 256), 256, 0, stream>>>(xA, xA_bf, XSZ);
    wt_kernel<<<dim3(VOCABN / 32, DIMN / 32), 256, 0, stream>>>(logits_W, lWt, DIMN, VOCABN);
    gemm_kernel<1><<<dim3(VOCABN / 64, ROWS / 128), 256, 0, stream>>>(
        xA_bf, lWt, logits_b, nullptr, out_logits, nullptr, ROWS, VOCABN, DIMN);
}
